// DeeperGCN_79474074845284
// MI455X (gfx1250) — matrix operand order repaired and checked by reading, not compiled, not run on hardware
//
#include <hip/hip_runtime.h>
#include <hip/hip_bf16.h>
#include <math.h>

#define HDIM 128
#define H2DIM 256

typedef __attribute__((ext_vector_type(16))) __bf16 v16bf;
typedef __attribute__((ext_vector_type(2)))  __bf16 v2bf;
typedef __attribute__((ext_vector_type(8)))  float  v8f;
typedef __attribute__((ext_vector_type(4)))  unsigned gu4;
typedef __attribute__((ext_vector_type(8)))  int      gi8;
typedef __attribute__((ext_vector_type(4)))  int      gi4;

struct U8  { unsigned u[8]; };   // 32B -> v16bf
struct Q2  { uint4 q[2]; };      // 32B -> v16bf

#if __has_builtin(__builtin_amdgcn_tensor_load_to_lds) && \
    __has_builtin(__builtin_amdgcn_s_wait_tensorcnt)
#define USE_TDM 1
#else
#define USE_TDM 0
#endif

// pack two fp32 -> two bf16 in one dword (round-half-up + v_perm_b32).
static __device__ __forceinline__ unsigned pk_bf16(float lo, float hi) {
#if __has_builtin(__builtin_amdgcn_cvt_pk_bf16_f32)
    v2bf p = __builtin_amdgcn_cvt_pk_bf16_f32(lo, hi);
    return __builtin_bit_cast(unsigned, p);
#else
    unsigned ul = __builtin_bit_cast(unsigned, lo) + 0x8000u;
    unsigned uh = __builtin_bit_cast(unsigned, hi) + 0x8000u;
    return __builtin_amdgcn_perm(uh, ul, 0x07060302u);
#endif
}

// ---------------------------------------------------------------------------
// Zero-fill
// ---------------------------------------------------------------------------
__global__ void memset_f32(float* __restrict__ p, long long n) {
    long long i = (long long)blockIdx.x * blockDim.x + threadIdx.x;
    long long stride = (long long)gridDim.x * blockDim.x;
    for (; i < n; i += stride) p[i] = 0.0f;
}

// ---------------------------------------------------------------------------
// Weight prep: WT[b][n][k] = bf16(W[b][k][n])  (RNE; off the hot path)
// ---------------------------------------------------------------------------
__global__ void transpose_bf16(const float* __restrict__ W,
                               unsigned short* __restrict__ WT,
                               int K, int Ncols, int nmat) {
    long long total = (long long)nmat * K * Ncols;
    long long i = (long long)blockIdx.x * blockDim.x + threadIdx.x;
    long long stride = (long long)gridDim.x * blockDim.x;
    const long long mm = (long long)K * Ncols;
    for (; i < total; i += stride) {
        long long b = i / mm;
        int rem = (int)(i - b * mm);
        int k = rem / Ncols;
        int n = rem - k * Ncols;
        unsigned u = __builtin_bit_cast(unsigned, W[i]);
        u = u + 0x7FFFu + ((u >> 16) & 1u);
        WT[b * mm + (long long)n * K + k] = (unsigned short)(u >> 16);
    }
}

// ---------------------------------------------------------------------------
// Edge message + scatter-add: agg[dst] += relu(feat[src] + Etab[ea]) + eps
// ---------------------------------------------------------------------------
__global__ __launch_bounds__(256) void edge_scatter(
    const float* __restrict__ feat, const int* __restrict__ src,
    const int* __restrict__ dst, const int* __restrict__ ea,
    const float* __restrict__ Etab, float* __restrict__ agg, int E)
{
    long long idx = (long long)blockIdx.x * blockDim.x + threadIdx.x;
    int e = (int)(idx >> 5);
    if (e >= E) return;
    int c = (int)(idx & 31) * 4;
    int s = src[e], d = dst[e], a = ea[e];
    const float4 hv = *reinterpret_cast<const float4*>(feat + (size_t)s * HDIM + c);
    const float4 ev = *reinterpret_cast<const float4*>(Etab + (size_t)a * HDIM + c);
    float m0 = fmaxf(hv.x + ev.x, 0.0f) + 1e-7f;
    float m1 = fmaxf(hv.y + ev.y, 0.0f) + 1e-7f;
    float m2 = fmaxf(hv.z + ev.z, 0.0f) + 1e-7f;
    float m3 = fmaxf(hv.w + ev.w, 0.0f) + 1e-7f;
    float* ap = agg + (size_t)d * HDIM + c;
    unsafeAtomicAdd(ap + 0, m0);
    unsafeAtomicAdd(ap + 1, m1);
    unsafeAtomicAdd(ap + 2, m2);
    unsafeAtomicAdd(ap + 3, m3);
}

// ---------------------------------------------------------------------------
// WMMA GEMM: C[M x NCOLS] = (A (+A2)) [M x K] @ B [K x NCOLS] + bias (+ Res)
// B (bf16, pre-transposed [NCOLS x K]) is staged per 128-K panel into LDS by
// the Tensor Data Mover (pad_amount inserts +16B/row for bank-conflict-free
// ds_load_b128 fragments). 8 waves x (16 rows x 128 cols); K fully unrolled.
// ---------------------------------------------------------------------------
template<int K, int NCOLS, bool HAS_A2, bool HAS_RES>
__global__ __launch_bounds__(256) void wmma_gemm(
    const float* __restrict__ A, const float* __restrict__ A2,
    const unsigned short* __restrict__ BT, const float* __restrict__ bias,
    const float* __restrict__ Res, float* __restrict__ C, int M)
{
    constexpr int KB = 128;                 // staged K-panel
    constexpr int LDS_STRIDE = KB + 8;      // +16B pad per row
    __shared__ unsigned short sB[128 * LDS_STRIDE];

    const int lane   = threadIdx.x & 31;
    const int wave   = threadIdx.x >> 5;
    const int lane16 = lane & 15;
    const int lhalf  = lane >> 4;

    const int m0 = blockIdx.x * 128 + wave * 16;
    const int n0 = blockIdx.y * 128;

    // Clamp A row on the M tail (keeps EXEC all-ones for WMMA; clamped rows
    // are computed but never stored).
    int arow = m0 + lane16;
    if (arow >= M) arow = M - 1;
    const float* __restrict__ Ab  = A + (size_t)arow * K + 16 * lhalf;
    const float* __restrict__ A2b = HAS_A2 ? (A2 + (size_t)arow * K + 16 * lhalf)
                                           : nullptr;
    const unsigned short* __restrict__ sBb =
        sB + lane16 * LDS_STRIDE + 16 * lhalf;

    v8f acc[8];
#pragma unroll
    for (int t = 0; t < 8; ++t)
#pragma unroll
        for (int j = 0; j < 8; ++j) acc[t][j] = 0.0f;

#pragma unroll
    for (int kk = 0; kk < K; kk += KB) {
        if (kk != 0) __syncthreads();       // previous panel fully consumed

        // ---- stage B panel rows [n0, n0+128) x K [kk, kk+128) into LDS ----
        const unsigned short* gB = BT + (size_t)n0 * K + kk;
#if USE_TDM
        if (threadIdx.x < 32) {             // one wave issues the DMA
            const unsigned lds_off = (unsigned)(size_t)(void*)&sB[0];
            const unsigned long long ga = (unsigned long long)(size_t)gB;
            gu4 g0;
            g0[0] = 1u;                                   // count=1
            g0[1] = lds_off;                              // lds_addr (bytes)
            g0[2] = (unsigned)ga;                         // global_addr lo
            g0[3] = (unsigned)((ga >> 32) & 0x01FFFFFFu)  // global_addr hi
                  | (2u << 30);                           // type=2 (image)
            gi8 g1;
            g1[0] = (int)((1u << 16)      // data_size = 2B
                        | (1u << 20)      // pad_enable
                        | (5u << 22)      // pad_interval: 64 DWORDs (=1 row)
                        | (3u << 25));    // pad_amount: 4 DWORDs (16B)
            g1[1] = (int)(128u << 16);    // tensor_dim0 = 128
            g1[2] = (int)(128u << 16);    // tensor_dim1 = 128
            g1[3] = (int)(128u << 16);    // tile_dim0 = 128
            g1[4] = 128;                  // tile_dim1 = 128
            g1[5] = K;                    // tensor_dim0_stride = K elems
            g1[6] = 0;
            g1[7] = 0;
            gi4 z4 = {0, 0, 0, 0};
            gi8 z8 = {0, 0, 0, 0, 0, 0, 0, 0};
            __builtin_amdgcn_tensor_load_to_lds(g0, g1, z4, z4, z8, 0);
            __builtin_amdgcn_s_wait_tensorcnt(0);
        }
#else
        for (int ch = threadIdx.x; ch < 128 * (KB / 8); ch += 256) {
            const int n = ch >> 4;          // KB/8 == 16 chunks per row
            const int c = ch & 15;
            *reinterpret_cast<uint4*>(&sB[n * LDS_STRIDE + c * 8]) =
                *reinterpret_cast<const uint4*>(gB + (size_t)n * K + c * 8);
        }
#endif
        __syncthreads();

        // ---- compute over the staged panel ----
#pragma unroll
        for (int k0 = 0; k0 < KB; k0 += 32) {
            // A fragment: 16 consecutive K per lane (k = 16*lhalf + s), matching B
            const int ka = kk + k0;
            float4 a0 = *reinterpret_cast<const float4*>(Ab + ka);
            float4 a1 = *reinterpret_cast<const float4*>(Ab + ka + 4);
            float4 a2 = *reinterpret_cast<const float4*>(Ab + ka + 8);
            float4 a3 = *reinterpret_cast<const float4*>(Ab + ka + 12);
            if constexpr (HAS_A2) {
                const float4 s0 = *reinterpret_cast<const float4*>(A2b + ka);
                const float4 s1 = *reinterpret_cast<const float4*>(A2b + ka + 4);
                const float4 s2 = *reinterpret_cast<const float4*>(A2b + ka + 8);
                const float4 s3 = *reinterpret_cast<const float4*>(A2b + ka + 12);
                a0.x += s0.x; a0.y += s0.y; a0.z += s0.z; a0.w += s0.w;
                a1.x += s1.x; a1.y += s1.y; a1.z += s1.z; a1.w += s1.w;
                a2.x += s2.x; a2.y += s2.y; a2.z += s2.z; a2.w += s2.w;
                a3.x += s3.x; a3.y += s3.y; a3.z += s3.z; a3.w += s3.w;
            }
            U8 au;
            au.u[0] = pk_bf16(a0.x, a0.y); au.u[1] = pk_bf16(a0.z, a0.w);
            au.u[2] = pk_bf16(a1.x, a1.y); au.u[3] = pk_bf16(a1.z, a1.w);
            au.u[4] = pk_bf16(a2.x, a2.y); au.u[5] = pk_bf16(a2.z, a2.w);
            au.u[6] = pk_bf16(a3.x, a3.y); au.u[7] = pk_bf16(a3.z, a3.w);
            const v16bf afrag = __builtin_bit_cast(v16bf, au);

            // B fragments from LDS: 16 consecutive K per lane, padded rows.
#pragma unroll
            for (int t = 0; t < 8; ++t) {
                const unsigned short* Bp = sBb + (t * 16 * LDS_STRIDE + k0);
                Q2 bq;
                bq.q[0] = *reinterpret_cast<const uint4*>(Bp);
                bq.q[1] = *reinterpret_cast<const uint4*>(Bp + 8);
                const v16bf bfrag = __builtin_bit_cast(v16bf, bq);
                acc[t] = __builtin_amdgcn_wmma_f32_16x16x32_bf16(
                    false, afrag, false, bfrag, (short)0, acc[t], false, false);
            }
        }
    }

    // --- Store: C/D layout — VGPR j, lane l: row = j + 8*(l>>4), col = l&15
    const int rbase = m0 + 8 * lhalf;
    float* __restrict__ Cb = C + (size_t)rbase * NCOLS + n0 + lane16;
    const float* __restrict__ Rb =
        HAS_RES ? (Res + (size_t)rbase * NCOLS + n0 + lane16) : nullptr;
    float bv[8];
#pragma unroll
    for (int t = 0; t < 8; ++t) bv[t] = bias[n0 + t * 16 + lane16];
#pragma unroll
    for (int j = 0; j < 8; ++j) {
        if (rbase + j < M) {
#pragma unroll
            for (int t = 0; t < 8; ++t) {
                float v = acc[t][j] + bv[t];
                if constexpr (HAS_RES) v += Rb[(size_t)j * NCOLS + t * 16];
                Cb[(size_t)j * NCOLS + t * 16] = v;
            }
        }
    }
}

// ---------------------------------------------------------------------------
// LayerNorm (+ optional ReLU), one wave32 per row. cols in {128, 256}.
// ---------------------------------------------------------------------------
__global__ __launch_bounds__(256) void ln_relu_kernel(
    const float* __restrict__ X, const float* __restrict__ g,
    const float* __restrict__ b, float* __restrict__ Y,
    int rows, int cols, int do_relu)
{
    const int wave = threadIdx.x >> 5;
    const int lane = threadIdx.x & 31;
    const int row = blockIdx.x * 8 + wave;
    if (row >= rows) return;
    const int per = cols >> 5;
    const float* __restrict__ x = X + (size_t)row * cols;
    float vals[8];
    float s = 0.0f;
    for (int i = 0; i < per; ++i) {
        vals[i] = x[lane * per + i];
        s += vals[i];
    }
#pragma unroll
    for (int off = 16; off > 0; off >>= 1) s += __shfl_xor(s, off, 32);
    const float mean = s / (float)cols;
    float v = 0.0f;
    for (int i = 0; i < per; ++i) {
        float d = vals[i] - mean;
        v += d * d;
    }
#pragma unroll
    for (int off = 16; off > 0; off >>= 1) v += __shfl_xor(v, off, 32);
    const float rstd = rsqrtf(v / (float)cols + 1e-5f);
    float* __restrict__ y = Y + (size_t)row * cols;
    for (int i = 0; i < per; ++i) {
        const int c = lane * per + i;
        float o = g[c] * (vals[i] - mean) * rstd + b[c];
        if (do_relu) o = fmaxf(o, 0.0f);
        y[c] = o;
    }
}

// ---------------------------------------------------------------------------
// Global mean pool: atomic accumulation of sums[G,128] and counts[G]
// ---------------------------------------------------------------------------
__global__ __launch_bounds__(256) void pool_accum(
    const float* __restrict__ h, const int* __restrict__ batch,
    float* __restrict__ sums, float* __restrict__ cnt, int N)
{
    long long idx = (long long)blockIdx.x * blockDim.x + threadIdx.x;
    int n = (int)(idx >> 5);
    if (n >= N) return;
    int sub = (int)(idx & 31);
    int c = sub * 4;
    int bg = batch[n];
    const float4 hv = *reinterpret_cast<const float4*>(h + (size_t)n * HDIM + c);
    float* sp = sums + (size_t)bg * HDIM + c;
    unsafeAtomicAdd(sp + 0, hv.x);
    unsafeAtomicAdd(sp + 1, hv.y);
    unsafeAtomicAdd(sp + 2, hv.z);
    unsafeAtomicAdd(sp + 3, hv.w);
    if (sub == 0) unsafeAtomicAdd(&cnt[bg], 1.0f);
}

// ---------------------------------------------------------------------------
// Head: out[g] = sigmoid((sums[g]/max(cnt,1)) . Wp + bp); one wave per graph
// ---------------------------------------------------------------------------
__global__ __launch_bounds__(256) void head_kernel(
    const float* __restrict__ sums, const float* __restrict__ cnt,
    const float* __restrict__ Wp, const float* __restrict__ bp,
    float* __restrict__ out, int G)
{
    const int wave = threadIdx.x >> 5;
    const int lane = threadIdx.x & 31;
    const int g = blockIdx.x * 8 + wave;
    if (g >= G) return;
    const float inv = 1.0f / fmaxf(cnt[g], 1.0f);
    float dot = 0.0f;
#pragma unroll
    for (int i = 0; i < 4; ++i) {
        const int k = lane * 4 + i;
        dot += sums[(size_t)g * HDIM + k] * inv * Wp[k];
    }
#pragma unroll
    for (int off = 16; off > 0; off >>= 1) dot += __shfl_xor(dot, off, 32);
    if (lane == 0) out[g] = 1.0f / (1.0f + expf(-(dot + bp[0])));
}

// ---------------------------------------------------------------------------
// Launcher
// ---------------------------------------------------------------------------
extern "C" void kernel_launch(void* const* d_in, const int* in_sizes, int n_in,
                              void* d_out, int out_size, void* d_ws, size_t ws_size,
                              hipStream_t stream) {
    const float* x    = (const float*)d_in[0];
    const int*   ei   = (const int*)d_in[1];      // [2, E]
    const int*   ea   = (const int*)d_in[2];      // [E]
    const int*   batch= (const int*)d_in[3];      // [N]
    const float* We   = (const float*)d_in[4];
    const float* be   = (const float*)d_in[5];
    const float* Etab = (const float*)d_in[6];
    const float* W1   = (const float*)d_in[7];    // [L,128,256]
    const float* b1   = (const float*)d_in[8];    // [L,256]
    const float* g1   = (const float*)d_in[9];
    const float* bb1  = (const float*)d_in[10];
    const float* W2   = (const float*)d_in[11];   // [L,256,128]
    const float* b2   = (const float*)d_in[12];   // [L,128]
    const float* gn   = (const float*)d_in[13];
    const float* bn   = (const float*)d_in[14];
    const float* Wp   = (const float*)d_in[15];   // [128]
    const float* bp   = (const float*)d_in[16];   // [1]

    const int E = in_sizes[2];
    const int N = in_sizes[3];
    const int G = out_size;
    const int L = in_sizes[8] / H2DIM;            // = 4
    const int* src = ei;
    const int* dst = ei + E;

    const size_t NH = (size_t)N * HDIM;
    float* h    = (float*)d_ws;
    float* h2   = h + NH;
    float* agg  = h2 + NH;
    float* y    = agg + NH;                       // N x 256
    float* sums = y + 2 * NH;                     // G x 128
    float* cnt  = sums + (size_t)G * HDIM;        // G

    // bf16 transposed weights (64B-aligned region after cnt)
    size_t wofs = 5 * NH + (size_t)G * HDIM + (size_t)G;
    wofs = (wofs + 15) & ~(size_t)15;
    unsigned short* WeT = (unsigned short*)((float*)d_ws + wofs);
    unsigned short* W1T = WeT + (size_t)HDIM * HDIM;
    unsigned short* W2T = W1T + (size_t)L * HDIM * H2DIM;

    const dim3 blk(256);
    const dim3 gemm1_grid((N + 127) / 128, 2);    // NCOLS=256
    const dim3 gemm2_grid((N + 127) / 128, 1);    // NCOLS=128
    const int  ln_grid = (N + 7) / 8;

    // weight prep (tiny; once per launch, deterministic)
    transpose_bf16<<<64,  blk, 0, stream>>>(We, WeT, HDIM, HDIM, 1);
    transpose_bf16<<<256, blk, 0, stream>>>(W1, W1T, HDIM, H2DIM, L);
    transpose_bf16<<<256, blk, 0, stream>>>(W2, W2T, H2DIM, HDIM, L);

    // node encoder: h = x @ We + be
    wmma_gemm<HDIM, HDIM, false, false><<<gemm2_grid, blk, 0, stream>>>(
        x, nullptr, WeT, be, nullptr, h, N);

    for (int l = 0; l < L; ++l) {
        const float* feat = (l == 0) ? h : h2;
        memset_f32<<<4096, blk, 0, stream>>>(agg, (long long)NH);
        edge_scatter<<<(int)(((long long)E * 32 + 255) / 256), blk, 0, stream>>>(
            feat, src, dst, ea, Etab, agg, E);
        // y = (feat + agg) @ W1[l] + b1[l]
        wmma_gemm<HDIM, H2DIM, true, false><<<gemm1_grid, blk, 0, stream>>>(
            feat, agg, W1T + (size_t)l * HDIM * H2DIM, b1 + (size_t)l * H2DIM,
            nullptr, y, N);
        // y = relu(LN(y))
        ln_relu_kernel<<<ln_grid, blk, 0, stream>>>(
            y, g1 + (size_t)l * H2DIM, bb1 + (size_t)l * H2DIM, y, N, H2DIM, 1);
        // h = y @ W2[l] + b2[l] (+ residual h for l>0)
        if (l == 0) {
            wmma_gemm<H2DIM, HDIM, false, false><<<gemm2_grid, blk, 0, stream>>>(
                y, nullptr, W2T + (size_t)l * H2DIM * HDIM, b2 + (size_t)l * HDIM,
                nullptr, h, N);
        } else {
            wmma_gemm<H2DIM, HDIM, false, true><<<gemm2_grid, blk, 0, stream>>>(
                y, nullptr, W2T + (size_t)l * H2DIM * HDIM, b2 + (size_t)l * HDIM,
                h, h, N);
        }
        if (l < L - 1) {
            ln_relu_kernel<<<ln_grid, blk, 0, stream>>>(
                h, gn + (size_t)l * HDIM, bn + (size_t)l * HDIM, h2, N, HDIM, 1);
        }
    }
    // final LN (no relu) -> h2
    ln_relu_kernel<<<ln_grid, blk, 0, stream>>>(
        h, gn + (size_t)(L - 1) * HDIM, bn + (size_t)(L - 1) * HDIM, h2, N, HDIM, 0);

    // global mean pool + sigmoid head
    memset_f32<<<64, blk, 0, stream>>>(sums, (long long)G * HDIM + G);
    pool_accum<<<(int)(((long long)N * 32 + 255) / 256), blk, 0, stream>>>(
        h2, batch, sums, cnt, N);
    head_kernel<<<(G + 7) / 8, blk, 0, stream>>>(
        sums, cnt, Wp, bp, (float*)d_out, G);
}
